// Block_21157008900513
// MI455X (gfx1250) — matrix operand order repaired and checked by reading, not compiled, not run on hardware
//
#include <hip/hip_runtime.h>
#include <hip/hip_bf16.h>

typedef __attribute__((ext_vector_type(16))) __bf16 v16bf;
typedef __attribute__((ext_vector_type(8)))  float  v8f;
typedef __attribute__((ext_vector_type(4)))  unsigned v4u;
typedef __attribute__((ext_vector_type(8)))  int    v8i_t;
typedef __attribute__((ext_vector_type(4)))  int    v4i_t;
typedef unsigned short u16;

#define BB   4
#define TT   2048
#define CC   1024
#define HH   16
#define DD   64
#define C3   3072
#define FFD  4096
#define NTOK 8192

#if __has_builtin(__builtin_amdgcn_tensor_load_to_lds)
#define USE_TDM 1
#else
#define USE_TDM 0
#endif

__device__ __forceinline__ u16 f2b(float f) {
  __bf16 h = (__bf16)f;
  return __builtin_bit_cast(u16, h);
}

union Frag {
  uint4 u[2];
  v16bf v;
};

__device__ __forceinline__ v8f wmma_bf16(v16bf a, v16bf b, v8f c) {
  // D = A(16x32) * B(32x16) + C, fp32 accumulate
  return __builtin_amdgcn_wmma_f32_16x16x32_bf16(false, a, false, b, (short)0, c,
                                                 false, false);
}

#if USE_TDM
// Issue one TDM load of a 128(rows) x 32(elems) bf16 tile into LDS.
// D# group0/group1 per CDNA5 ISA §8.3/8.4; groups 2/3 unused (2D tensor).
// This toolchain exposes the 6-arg builtin:
//   (uint32x4 g0, int32x8 g1, int32x4 g2, int32x4 g3, int32x8 pad, i32 cpol)
__device__ __forceinline__ void tdm_load_tile_128x32(unsigned lds_addr,
                                                     const u16* gptr,
                                                     int rows_tensor, int K) {
  const unsigned long long ga = (unsigned long long)(uintptr_t)gptr;
  v4u g0;
  g0[0] = 1u;                                   // count=1, user descriptor
  g0[1] = lds_addr;                             // LDS byte address
  g0[2] = (unsigned)(ga & 0xffffffffu);         // global_addr[31:0]
  g0[3] = (unsigned)((ga >> 32) & 0x1ffffffu)   // global_addr[56:32]
          | (2u << 30);                         // type = image
  v8i_t g1;
  g1[0] = (int)(1u << 16);                      // wg_mask=0, data_size=1 (2B)
  g1[1] = (int)(((unsigned)K & 0xffffu) << 16); // tensor_dim0[15:0]
  g1[2] = (int)((((unsigned)K) >> 16) |
                (((unsigned)rows_tensor & 0xffffu) << 16)); // dim0 hi / dim1 lo
  g1[3] = (int)((((unsigned)rows_tensor) >> 16) |
                (32u << 16));                   // dim1 hi / tile_dim0 = 32
  g1[4] = 128;                                  // tile_dim1 = 128, tile_dim2 = 0
  g1[5] = K;                                    // tensor_dim0_stride[31:0]
  g1[6] = 0;                                    // stride hi / dim1_stride lo
  g1[7] = 0;                                    // dim1_stride hi
  const v4i_t zz4 = {0, 0, 0, 0};
  const v8i_t zz8 = {0, 0, 0, 0, 0, 0, 0, 0};
  __builtin_amdgcn_tensor_load_to_lds(g0, g1, zz4, zz4, zz8, 0);
}
#endif

// ---------------------------------------------------------------- fp32->bf16
__global__ __launch_bounds__(256) void cvt_f32_bf16(const float* __restrict__ in,
                                                    u16* __restrict__ out, int n) {
  int i = blockIdx.x * 256 + threadIdx.x;
  if (i < n) out[i] = f2b(in[i]);
}

// -------------------------------------------------- LayerNorm (fused ->bf16)
__global__ __launch_bounds__(256) void layernorm_bf16_k(const float* __restrict__ x,
                                                        const float* __restrict__ g,
                                                        u16* __restrict__ out) {
  const int row = blockIdx.x;
  const float* xr = x + (size_t)row * CC;
  const int tid = threadIdx.x;
  float4 v = *(const float4*)(xr + tid * 4);
  float s  = v.x + v.y + v.z + v.w;
  float ss = v.x * v.x + v.y * v.y + v.z * v.z + v.w * v.w;
#pragma unroll
  for (int m = 16; m >= 1; m >>= 1) {
    s  += __shfl_xor(s, m, 32);
    ss += __shfl_xor(ss, m, 32);
  }
  __shared__ float sm[8], sq[8];
  if ((tid & 31) == 0) { sm[tid >> 5] = s; sq[tid >> 5] = ss; }
  __syncthreads();
  float ts = 0.f, tq = 0.f;
#pragma unroll
  for (int w = 0; w < 8; ++w) { ts += sm[w]; tq += sq[w]; }
  const float mean = ts * (1.0f / CC);
  const float var  = tq * (1.0f / CC) - mean * mean;
  const float rstd = rsqrtf(var + 1e-5f);
  float4 gv = *(const float4*)(g + tid * 4);
  u16* orow = out + (size_t)row * CC + tid * 4;
  orow[0] = f2b((v.x - mean) * rstd * gv.x);
  orow[1] = f2b((v.y - mean) * rstd * gv.y);
  orow[2] = f2b((v.z - mean) * rstd * gv.z);
  orow[3] = f2b((v.w - mean) * rstd * gv.w);
}

// ------------------------------------------------------------- bf16 WMMA GEMM
// C[M,N] = A[M,K] * Bw[N,K]^T  (+ GELU) (+ fp32 residual), out bf16 or fp32.
// 256 threads = 8 waves; wave grid 4(M) x 2(N); each wave: 2x4 tiles of 16x16.
// Tile staging: TDM tensor_load_to_lds (wave 0 issues, TENSORcnt-waited),
// double-buffered so DMA of step kt+1 overlaps WMMA of step kt.
template <bool OUT_BF16, bool GELU, bool RES>
__global__ __launch_bounds__(256) void gemm_bf16_k(const u16* __restrict__ A,
                                                   const u16* __restrict__ Bw,
                                                   void* __restrict__ Cout,
                                                   const float* __restrict__ Res,
                                                   int M, int N, int K) {
  __shared__ u16 As[2][128 * 32];
  __shared__ u16 Bs[2][128 * 32];
  const int tid   = threadIdx.x;
  const int lane  = tid & 31;
  const int wid   = tid >> 5;
  const int lhi   = lane >> 4;
  const int llo   = lane & 15;
  const int waveM = wid & 3;
  const int waveN = wid >> 2;
  const size_t bM = (size_t)blockIdx.y * 128;
  const size_t bN = (size_t)blockIdx.x * 128;

  v8f acc[2][4];
#pragma unroll
  for (int mt = 0; mt < 2; ++mt)
#pragma unroll
    for (int nt = 0; nt < 4; ++nt)
      acc[mt][nt] = (v8f){0.f, 0.f, 0.f, 0.f, 0.f, 0.f, 0.f, 0.f};

#if USE_TDM
  auto issue_tiles = [&](int kb, int buf) {
    tdm_load_tile_128x32((unsigned)(uintptr_t)&As[buf][0], A + bM * K + kb, M, K);
    tdm_load_tile_128x32((unsigned)(uintptr_t)&Bs[buf][0], Bw + bN * K + kb, N, K);
  };
  if (wid == 0) {
    issue_tiles(0, 0);
    __builtin_amdgcn_s_wait_tensorcnt(0);
  }
#else
  const int row0 = tid >> 2;       // 0..63
  const int seg  = (tid & 3) * 8;  // bf16 element offset within a 32-elem row
  auto issue_tiles = [&](int kb, int buf) {
#pragma unroll
    for (int i = 0; i < 2; ++i) {
      const int r = row0 + i * 64;
      *(uint4*)(&As[buf][r * 32 + seg]) = *(const uint4*)(A + (bM + r) * K + kb + seg);
      *(uint4*)(&Bs[buf][r * 32 + seg]) = *(const uint4*)(Bw + (bN + r) * K + kb + seg);
    }
  };
  issue_tiles(0, 0);
#endif
  __syncthreads();

  const int nk = K / 32;
  for (int kt = 0; kt < nk; ++kt) {
    const int buf = kt & 1;
#if USE_TDM
    if (kt + 1 < nk && wid == 0) issue_tiles((kt + 1) * 32, buf ^ 1);
#else
    if (kt + 1 < nk) issue_tiles((kt + 1) * 32, buf ^ 1);
#endif

    Frag af[2], bf[4];
#pragma unroll
    for (int mt = 0; mt < 2; ++mt) {  // A frag: lane-lo K0-7/16-23, lane-hi K8-15/24-31
      const u16* p = &As[buf][(waveM * 32 + mt * 16 + llo) * 32 + lhi * 8];
      af[mt].u[0] = *(const uint4*)p;
      af[mt].u[1] = *(const uint4*)(p + 16);
    }
#pragma unroll
    for (int nt = 0; nt < 4; ++nt) {  // B frag: lane-lo K0-7/16-23, lane-hi K8-15/24-31
      const u16* p = &Bs[buf][(waveN * 64 + nt * 16 + llo) * 32 + lhi * 8];
      bf[nt].u[0] = *(const uint4*)p;
      bf[nt].u[1] = *(const uint4*)(p + 16);
    }
#pragma unroll
    for (int nt = 0; nt < 4; ++nt)
#pragma unroll
      for (int mt = 0; mt < 2; ++mt)
        acc[mt][nt] = wmma_bf16(af[mt].v, bf[nt].v, acc[mt][nt]);

#if USE_TDM
    if (wid == 0) __builtin_amdgcn_s_wait_tensorcnt(0);
#endif
    __syncthreads();
  }

#pragma unroll
  for (int mt = 0; mt < 2; ++mt) {
#pragma unroll
    for (int nt = 0; nt < 4; ++nt) {
      const size_t r0 = bM + waveM * 32 + mt * 16 + lhi * 8;
      const size_t c0 = bN + waveN * 64 + nt * 16 + llo;
#pragma unroll
      for (int i = 0; i < 8; ++i) {
        const size_t off = (r0 + i) * N + c0;
        float v = acc[mt][nt][i];
        if (GELU) v = 0.5f * v * (1.0f + erff(v * 0.70710678118654752f));
        if (RES) v += Res[off];
        if (OUT_BF16) ((u16*)Cout)[off] = f2b(v);
        else          ((float*)Cout)[off] = v;
      }
    }
  }
}

// -------------------------------------------- V transpose: qkv -> Vt[B,H,D,T]
__global__ __launch_bounds__(256) void vtrans_k(const u16* __restrict__ qkv,
                                                u16* __restrict__ vt) {
  const size_t i = (size_t)blockIdx.x * 256 + threadIdx.x;  // over B*T*C
  const int c  = (int)(i & (CC - 1));
  const size_t bt = i >> 10;        // b*T + t
  const int t = (int)(bt & (TT - 1));
  const int b = (int)(bt >> 11);
  vt[((size_t)(b * CC + c)) * TT + t] = qkv[bt * C3 + 2 * CC + c];
}

// ----------------------------------------- flash attention (causal, bf16 WMMA)
// grid = (T/128, B*H); 8 waves/block; wave w owns 16 query rows.
__global__ __launch_bounds__(256) void attn_k(const u16* __restrict__ qkv,
                                              const u16* __restrict__ vt,
                                              u16* __restrict__ y) {
  const int bh  = blockIdx.y;
  const int b   = bh >> 4, h = bh & 15;
  const int wid = threadIdx.x >> 5;
  const int lane = threadIdx.x & 31;
  const int lhi = lane >> 4, llo = lane & 15;
  const int qbase = blockIdx.x * 128 + wid * 16;

  const u16* qp = qkv + (size_t)b * TT * C3 + h * DD;
  const u16* kp = qp + CC;
  const u16* vp = vt + (size_t)bh * DD * TT;

  __shared__ u16 Pl[8][16 * 32];  // per-wave P scratch (C-layout -> A-layout)

  Frag qf[2];
  {
    const u16* qrow = qp + (size_t)(qbase + llo) * C3;
#pragma unroll
    for (int c = 0; c < 2; ++c) {
      qf[c].u[0] = *(const uint4*)(qrow + c * 32 + lhi * 8);
      qf[c].u[1] = *(const uint4*)(qrow + c * 32 + 16 + lhi * 8);
    }
  }

  const v8f vz = (v8f){0.f, 0.f, 0.f, 0.f, 0.f, 0.f, 0.f, 0.f};
  v8f acc[4] = {vz, vz, vz, vz};
  float mI[8], lI[8];
#pragma unroll
  for (int i = 0; i < 8; ++i) { mI[i] = -1e30f; lI[i] = 0.f; }

  const int kend = qbase + 16;  // causal: keys [0, kend)
  for (int kb = 0; kb < kend; kb += 32) {
    Frag kf[2][2];
#pragma unroll
    for (int t2 = 0; t2 < 2; ++t2) {
      const u16* krow = kp + (size_t)(kb + t2 * 16 + llo) * C3;
#pragma unroll
      for (int c = 0; c < 2; ++c) {
        kf[t2][c].u[0] = *(const uint4*)(krow + c * 32 + lhi * 8);
        kf[t2][c].u[1] = *(const uint4*)(krow + c * 32 + 16 + lhi * 8);
      }
    }
    v8f s0 = wmma_bf16(qf[0].v, kf[0][0].v, vz);
    s0 = wmma_bf16(qf[1].v, kf[0][1].v, s0);
    v8f s1 = wmma_bf16(qf[0].v, kf[1][0].v, vz);
    s1 = wmma_bf16(qf[1].v, kf[1][1].v, s1);

#pragma unroll
    for (int i = 0; i < 8; ++i) {
      const int row = qbase + i + 8 * lhi;
      const int j0 = kb + llo;
      const int j1 = kb + 16 + llo;
      float a0 = (j0 <= row) ? s0[i] * 0.125f : -1e30f;
      float a1 = (j1 <= row) ? s1[i] * 0.125f : -1e30f;
      float mx = fmaxf(a0, a1);
#pragma unroll
      for (int m = 8; m >= 1; m >>= 1) mx = fmaxf(mx, __shfl_xor(mx, m, 32));
      const float mnew = fmaxf(mI[i], mx);
      const float p0 = __expf(a0 - mnew);
      const float p1 = __expf(a1 - mnew);
      const float rs = __expf(mI[i] - mnew);
      float psum = p0 + p1;
#pragma unroll
      for (int m = 8; m >= 1; m >>= 1) psum += __shfl_xor(psum, m, 32);
      lI[i] = lI[i] * rs + psum;
      mI[i] = mnew;
#pragma unroll
      for (int nt = 0; nt < 4; ++nt) acc[nt][i] *= rs;
      const int prow = i + 8 * lhi;
      Pl[wid][prow * 32 + llo]      = f2b(p0);
      Pl[wid][prow * 32 + 16 + llo] = f2b(p1);
    }

    // wave-local cross-lane LDS round trip: drain DScnt before re-reading
    __builtin_amdgcn_wave_barrier();
    asm volatile("s_wait_dscnt 0" ::: "memory");

    Frag pf;
    {
      const u16* p = &Pl[wid][llo * 32 + lhi * 8];
      pf.u[0] = *(const uint4*)p;
      pf.u[1] = *(const uint4*)(p + 16);
    }
#pragma unroll
    for (int nt = 0; nt < 4; ++nt) {
      Frag vf;  // B frag from Vt: column d = nt*16+llo, keys 8h..8h+7 / 16+8h..
      const u16* vrow = vp + (size_t)(nt * 16 + llo) * TT + kb;
      vf.u[0] = *(const uint4*)(vrow + lhi * 8);
      vf.u[1] = *(const uint4*)(vrow + 16 + lhi * 8);
      acc[nt] = wmma_bf16(pf.v, vf.v, acc[nt]);
    }
    __builtin_amdgcn_wave_barrier();
  }

#pragma unroll
  for (int nt = 0; nt < 4; ++nt) {
#pragma unroll
    for (int i = 0; i < 8; ++i) {
      const int row = qbase + i + 8 * lhi;
      const float v = acc[nt][i] / lI[i];
      y[((size_t)(b * TT + row)) * CC + h * DD + nt * 16 + llo] = f2b(v);
    }
  }
}

// ---------------------------------------------------------------------- host
extern "C" void kernel_launch(void* const* d_in, const int* in_sizes, int n_in,
                              void* d_out, int out_size, void* d_ws, size_t ws_size,
                              hipStream_t stream) {
  (void)in_sizes; (void)n_in; (void)out_size; (void)ws_size;
  const float* x      = (const float*)d_in[0];
  const float* g1     = (const float*)d_in[1];
  const float* w_qkv  = (const float*)d_in[2];
  const float* w_o    = (const float*)d_in[3];
  const float* g2     = (const float*)d_in[4];
  const float* w_fc   = (const float*)d_in[5];
  const float* w_proj = (const float*)d_in[6];

  char* ws = (char*)d_ws;
  size_t off = 0;
  auto carve = [&](size_t bytes) -> char* {
    char* p = ws + off;
    off += (bytes + 255) & ~(size_t)255;
    return p;
  };
  u16*   wqkv_b  = (u16*)carve((size_t)3 * CC * CC * 2);
  u16*   wo_b    = (u16*)carve((size_t)CC * CC * 2);
  u16*   wfc_b   = (u16*)carve((size_t)FFD * CC * 2);
  u16*   wproj_b = (u16*)carve((size_t)CC * FFD * 2);
  u16*   xn_b    = (u16*)carve((size_t)NTOK * CC * 2);   // reused for ln1 & ln2
  u16*   qkv_b   = (u16*)carve((size_t)NTOK * C3 * 2);
  u16*   vt_b    = (u16*)carve((size_t)NTOK * CC * 2);
  u16*   y_b     = (u16*)carve((size_t)NTOK * CC * 2);
  float* x2      = (float*)carve((size_t)NTOK * CC * 4);
  u16*   h_b     = (u16*)carve((size_t)NTOK * FFD * 2);

  cvt_f32_bf16<<<(3 * CC * CC + 255) / 256, 256, 0, stream>>>(w_qkv, wqkv_b, 3 * CC * CC);
  cvt_f32_bf16<<<(CC * CC + 255) / 256, 256, 0, stream>>>(w_o, wo_b, CC * CC);
  cvt_f32_bf16<<<(FFD * CC + 255) / 256, 256, 0, stream>>>(w_fc, wfc_b, FFD * CC);
  cvt_f32_bf16<<<(CC * FFD + 255) / 256, 256, 0, stream>>>(w_proj, wproj_b, CC * FFD);

  layernorm_bf16_k<<<NTOK, 256, 0, stream>>>(x, g1, xn_b);

  gemm_bf16_k<true, false, false><<<dim3(C3 / 128, NTOK / 128), 256, 0, stream>>>(
      xn_b, wqkv_b, qkv_b, nullptr, NTOK, C3, CC);

  vtrans_k<<<(NTOK * CC) / 256, 256, 0, stream>>>(qkv_b, vt_b);

  attn_k<<<dim3(TT / 128, BB * HH), 256, 0, stream>>>(qkv_b, vt_b, y_b);

  gemm_bf16_k<false, false, true><<<dim3(CC / 128, NTOK / 128), 256, 0, stream>>>(
      y_b, wo_b, x2, x, NTOK, CC, CC);

  layernorm_bf16_k<<<NTOK, 256, 0, stream>>>(x2, g2, xn_b);

  gemm_bf16_k<true, true, false><<<dim3(FFD / 128, NTOK / 128), 256, 0, stream>>>(
      xn_b, wfc_b, h_b, nullptr, NTOK, FFD, CC);

  gemm_bf16_k<false, false, true><<<dim3(CC / 128, NTOK / 128), 256, 0, stream>>>(
      h_b, wproj_b, (float*)d_out, x2, NTOK, CC, FFD);
}
